// CTMDNXCell_76338748719529
// MI455X (gfx1250) — matrix operand order repaired and checked by reading, not compiled, not run on hardware
//
#include <hip/hip_runtime.h>
#include <hip/hip_bf16.h>

// ---------------- problem constants ----------------
#define BATCH   8192
#define INPUT   512
#define HIDDEN  1024
#define NSYNCH  1024
#define UNFOLDS 6
#define DELTA_T 0.1f
#define ADA_THR 0.01f
#define BH      (BATCH * HIDDEN)     // 8388608

typedef __attribute__((ext_vector_type(16))) __bf16        v16bf;
typedef __attribute__((ext_vector_type(8)))  float         v8f;
typedef __attribute__((ext_vector_type(4)))  unsigned int  u32x4;

// ---------------- CDNA5 async global->LDS path (probe-guarded) ----------------
#if defined(__AMDGCN__) && defined(__gfx1250__) && \
    __has_builtin(__builtin_amdgcn_global_load_async_to_lds_b128) && \
    __has_builtin(__builtin_amdgcn_s_wait_asynccnt)
#define USE_ASYNC_LDS 1
#else
#define USE_ASYNC_LDS 0
#endif

#if USE_ASYNC_LDS
// Builtin signature (from clang diagnostics): param0 = v4i __device__* (AS1),
// param1 = v4i __shared__* (AS3), then imm offset, imm cpol. The imm offset is
// added to BOTH the global and LDS address (ISA 08 §4.4), so one VGPR address
// pair covers a contiguous 32B move as two b128 ops.
typedef int v4i_g __attribute__((vector_size(16)));
#define ASYNC_B128(gp, lp, IMM)                                                    \
    __builtin_amdgcn_global_load_async_to_lds_b128(                                \
        (__attribute__((address_space(1))) v4i_g*)(unsigned long long)(gp),        \
        (__attribute__((address_space(3))) v4i_g*)(unsigned int)(unsigned long long)(lp), \
        IMM, 0)
#endif

__device__ inline void wait_async_lds() {
#if USE_ASYNC_LDS
    __builtin_amdgcn_s_wait_asynccnt(0);
#endif
}

// cooperative 128x32 bf16 tile stage: 256 threads, each moves 32 bytes
__device__ inline void stage_tile(const __bf16* __restrict__ g, int row0, int ld,
                                  int kc, __bf16 (*T)[32], int tid) {
    int row  = tid >> 1;
    int hsel = tid & 1;
    const __bf16* src = g + (size_t)(row0 + row) * ld + kc + hsel * 16;
    __bf16*       dst = &T[row][hsel * 16];
#if USE_ASYNC_LDS
    ASYNC_B128(src, dst, 0);
    ASYNC_B128(src, dst, 16);
#else
    ((u32x4*)dst)[0] = ((const u32x4*)src)[0];
    ((u32x4*)dst)[1] = ((const u32x4*)src)[1];
#endif
}

// ---------------- WMMA fragment gathers from LDS ----------------
// LDS tiles: T[row][32] bf16 (row = M for A tiles, row = N for B tiles), K-contiguous.

// A-matrix 16x32 bf16 layout (ISA 7.12.2):
//   lane<16 : M=lane,    elems 0..7 -> K=0..7,   elems 8..15 -> K=16..23
//   lane>=16: M=lane-16, elems 0..7 -> K=8..15,  elems 8..15 -> K=24..31
__device__ inline v16bf frag_a(const __bf16 (*T)[32], int rbase, int l16, int half) {
    union { v16bf v; u32x4 q[2]; } u;
    const __bf16* p = &T[rbase + l16][0];
    u.q[0] = *(const u32x4*)(p + half * 8);
    u.q[1] = *(const u32x4*)(p + 16 + half * 8);
    return u.v;
}

// B-matrix 32x16 bf16 layout: lane<16 -> N=lane holds K=0..7,16..23; lane>=16 -> K=8..15,24..31
__device__ inline v16bf frag_b(const __bf16 (*T)[32], int rbase, int l16, int half) {
    union { v16bf v; u32x4 q[2]; } u;
    const __bf16* p = &T[rbase + l16][0];
    u.q[0] = *(const u32x4*)(p + half * 8);
    u.q[1] = *(const u32x4*)(p + 16 + half * 8);
    return u.v;
}

// ---------------- fused, double-buffered WMMA GEMM ----------------
// C[8192,1024](f32) = A1(bf16)[8192,KTOT] x B1(bf16)[1024,KTOT]^T (+ A2 x B2^T if DUAL)
// STEP epilogue: v += x_drive + b_h; f=tanh(v); hu = dt*(f - h*inv_tau);
//                Cout=hu; atomic-accumulate sum |hu|.
template<int KTOT, bool DUAL, bool STEP>
__global__ __launch_bounds__(256) void k_gemm(
    const __bf16* __restrict__ A1, const __bf16* __restrict__ B1,
    const __bf16* __restrict__ A2, const __bf16* __restrict__ B2,
    const float*  __restrict__ addend,   // x_drive [B,H]     (STEP)
    const float*  __restrict__ bias,     // b_h [H]           (STEP)
    const float*  __restrict__ hstate,   // h f32 [B,H]       (STEP)
    float*        __restrict__ Cout,     // x_drive or h_update
    const float*  __restrict__ scal,     // [0]=inv_tau
    const int*    __restrict__ done,
    float*        __restrict__ accum)
{
    if (STEP) { if (*done) return; }     // uniform, graph-safe gate

    __shared__ __bf16 As1[2][128][32];
    __shared__ __bf16 Bs1[2][128][32];
    __shared__ __bf16 As2[DUAL ? 2 : 1][DUAL ? 128 : 1][32];
    __shared__ __bf16 Bs2[DUAL ? 2 : 1][DUAL ? 128 : 1][32];
    __shared__ float  red[256];

    const int tid  = threadIdx.x;
    const int wave = tid >> 5;
    const int lane = tid & 31;
    const int l16  = lane & 15;
    const int half = lane >> 4;
    const int wM   = wave & 3;     // 4 M-strips of 32 rows
    const int wN   = wave >> 2;    // 2 N-strips of 64 cols
    const int m0   = blockIdx.x * 128;
    const int n0   = blockIdx.y * 128;

    v8f acc[2][4];
#pragma unroll
    for (int mi = 0; mi < 2; ++mi)
#pragma unroll
        for (int ni = 0; ni < 4; ++ni) acc[mi][ni] = (v8f){};

    auto stage_all = [&](int kc, int b) {
        stage_tile(A1, m0, KTOT, kc, As1[b], tid);
        stage_tile(B1, n0, KTOT, kc, Bs1[b], tid);
        if (DUAL) {
            stage_tile(A2, m0, KTOT, kc, As2[b], tid);
            stage_tile(B2, n0, KTOT, kc, Bs2[b], tid);
        }
    };

    constexpr int NCH = KTOT / 32;
    stage_all(0, 0);                        // prologue fill, buffer 0

    for (int c = 0; c < NCH; ++c) {
        const int pb = c & 1;
        wait_async_lds();                   // my chunk-c async loads landed
        __syncthreads();                    // all waves' data visible; prev reads done
        if (c + 1 < NCH) stage_all((c + 1) * 32, pb ^ 1);   // overlap next fill

        {
            v16bf bf[4];
#pragma unroll
            for (int ni = 0; ni < 4; ++ni)
                bf[ni] = frag_b(Bs1[pb], wN * 64 + ni * 16, l16, half);
#pragma unroll
            for (int mi = 0; mi < 2; ++mi) {
                v16bf af = frag_a(As1[pb], wM * 32 + mi * 16, l16, half);
#pragma unroll
                for (int ni = 0; ni < 4; ++ni)
                    acc[mi][ni] = __builtin_amdgcn_wmma_f32_16x16x32_bf16(
                        false, af, false, bf[ni], (short)0, acc[mi][ni], false, false);
            }
        }
        if (DUAL) {
            v16bf bf[4];
#pragma unroll
            for (int ni = 0; ni < 4; ++ni)
                bf[ni] = frag_b(Bs2[pb], wN * 64 + ni * 16, l16, half);
#pragma unroll
            for (int mi = 0; mi < 2; ++mi) {
                v16bf af = frag_a(As2[pb], wM * 32 + mi * 16, l16, half);
#pragma unroll
                for (int ni = 0; ni < 4; ++ni)
                    acc[mi][ni] = __builtin_amdgcn_wmma_f32_16x16x32_bf16(
                        false, af, false, bf[ni], (short)0, acc[mi][ni], false, false);
            }
        }
    }

    // epilogue — C/D layout: elem r -> M = r + 8*half, N = l16
    float lsum = 0.f;
    const float inv_tau = STEP ? scal[0] : 0.f;
#pragma unroll
    for (int mi = 0; mi < 2; ++mi) {
#pragma unroll
        for (int ni = 0; ni < 4; ++ni) {
#pragma unroll
            for (int r = 0; r < 8; ++r) {
                int m = m0 + wM * 32 + mi * 16 + r + 8 * half;
                int n = n0 + wN * 64 + ni * 16 + l16;
                size_t idx = (size_t)m * HIDDEN + n;
                float v = acc[mi][ni][r];
                if (STEP) {
                    v += addend[idx] + bias[n];
                    float f  = tanhf(v);
                    float hu = DELTA_T * (f - hstate[idx] * inv_tau);
                    Cout[idx] = hu;
                    lsum += fabsf(hu);
                } else {
                    Cout[idx] = v;
                }
            }
        }
    }

    if (STEP) {
        red[tid] = lsum;
        __syncthreads();
        for (int s = 128; s > 0; s >>= 1) {
            if (tid < s) red[tid] += red[tid + s];
            __syncthreads();
        }
        if (tid == 0) atomicAdd(accum, red[0]);
    }
}

// ---------------- elementwise / gather kernels ----------------
__global__ __launch_bounds__(256) void k_cvt(const float* __restrict__ src,
                                             __bf16* __restrict__ dst, int n) {
    int i = blockIdx.x * 256 + threadIdx.x;
    if (i < n) dst[i] = (__bf16)src[i];
}

__global__ __launch_bounds__(256) void k_init(
    const float* __restrict__ h_in, const float* __restrict__ a_in,
    const float* __restrict__ b_in, const float* __restrict__ taup,
    float* __restrict__ out, float* __restrict__ scal,
    int* __restrict__ done, float* __restrict__ accum)
{
    int i = blockIdx.x * 256 + threadIdx.x;
    if (i < BH) {
        out[i]              = h_in[i];
        out[(size_t)BH + i] = a_in[i];
        out[2*(size_t)BH+i] = b_in[i];
    }
    if (i == 0) {
        float tau = log1pf(expf(taup[0])) + 0.01f;   // softplus + 0.01
        scal[0] = 1.f / tau;   // inv_tau
        scal[1] = 2.f;         // apply scale
        *done   = 0;
        *accum  = 0.f;
        out[3 * (size_t)BH] = 0.f;                    // steps
    }
}

// per-element: act gathers, alpha/beta EMA, sync value (bf16), h->bf16 copy
__global__ __launch_bounds__(256) void k_sync(
    const float* __restrict__ h, const int* __restrict__ idxL,
    const int* __restrict__ idxR, const float* __restrict__ rparam,
    float* __restrict__ alpha, float* __restrict__ beta,
    __bf16* __restrict__ sync_bf, __bf16* __restrict__ h_bf,
    const int* __restrict__ done)
{
    if (*done) return;
    int i = blockIdx.x * 256 + threadIdx.x;           // i < B*NSYNCH
    int b = i >> 10;
    int s = i & (NSYNCH - 1);
    float rs = 1.f / (1.f + expf(-rparam[s]));
    float aL = tanhf(h[(size_t)b * HIDDEN + idxL[s]]);
    float aR = tanhf(h[(size_t)b * HIDDEN + idxR[s]]);
    float na = rs * alpha[i] + aL * aR;
    float nb = rs * beta[i] + 1.f;
    alpha[i] = na;
    beta[i]  = nb;
    sync_bf[i] = (__bf16)(na / (sqrtf(nb) + 1e-6f));
    h_bf[i]    = (__bf16)h[i];                         // NSYNCH == HIDDEN
}

__global__ void k_decide(float* __restrict__ accum, int* __restrict__ done,
                         float* __restrict__ scal, float* __restrict__ steps_out,
                         int step)
{
    if (threadIdx.x == 0 && blockIdx.x == 0) {
        if (!*done) {
            float mean = *accum / (float)BH;
            int brk = (step >= 3) && (mean < ADA_THR);
            scal[1] = brk ? 1.f : 2.f;      // break path applies once, else twice
            steps_out[0] = (float)step;
            if (brk) *done = 1;
        } else {
            scal[1] = 0.f;
        }
        *accum = 0.f;
    }
}

__global__ __launch_bounds__(256) void k_apply(float* __restrict__ h,
                                               const float* __restrict__ hu,
                                               const float* __restrict__ scal)
{
    int i = blockIdx.x * 256 + threadIdx.x;
    float s = scal[1];
    if (s != 0.f) h[i] += s * hu[i];
}

// ---------------- launch ----------------
extern "C" void kernel_launch(void* const* d_in, const int* in_sizes, int n_in,
                              void* d_out, int out_size, void* d_ws, size_t ws_size,
                              hipStream_t stream)
{
    const float* x     = (const float*)d_in[0];
    const float* h_in  = (const float*)d_in[1];
    const float* a_in  = (const float*)d_in[2];
    const float* b_in  = (const float*)d_in[3];
    const float* Wx    = (const float*)d_in[4];
    const float* Wh    = (const float*)d_in[5];
    const float* bh    = (const float*)d_in[6];
    const float* taup  = (const float*)d_in[7];
    const float* rpar  = (const float*)d_in[8];
    const float* Wm    = (const float*)d_in[9];
    const int*   idxL  = (const int*)d_in[10];
    const int*   idxR  = (const int*)d_in[11];

    float* out    = (float*)d_out;
    float* hstate = out;                       // live h
    float* alpha  = out + (size_t)BH;          // live alpha
    float* beta   = out + 2 * (size_t)BH;      // live beta

    char* ws = (char*)d_ws;
    size_t off = 0;
    float*  x_drive = (float*) (ws + off); off += (size_t)BH * 4;        // 32 MB
    float*  h_upd   = (float*) (ws + off); off += (size_t)BH * 4;        // 32 MB
    __bf16* h_bf    = (__bf16*)(ws + off); off += (size_t)BH * 2;        // 16 MB
    __bf16* sync_bf = (__bf16*)(ws + off); off += (size_t)BH * 2;        // 16 MB
    __bf16* x_bf    = (__bf16*)(ws + off); off += (size_t)BATCH*INPUT*2; //  8 MB
    __bf16* Wx_bf   = (__bf16*)(ws + off); off += (size_t)HIDDEN*INPUT*2;
    __bf16* Wh_bf   = (__bf16*)(ws + off); off += (size_t)HIDDEN*HIDDEN*2;
    __bf16* Wm_bf   = (__bf16*)(ws + off); off += (size_t)HIDDEN*NSYNCH*2;
    float*  scal    = (float*) (ws + off); off += 16;   // [0]=inv_tau [1]=apply_scale
    int*    done    = (int*)   (ws + off); off += 16;
    float*  accum   = (float*) (ws + off); off += 16;

    const dim3 gemm_grid(BATCH / 128, HIDDEN / 128);   // 64 x 8

    // bf16 conversions (weights are [N,K] row-major == WMMA B orientation)
    k_cvt<<<(HIDDEN*INPUT  + 255)/256, 256, 0, stream>>>(Wx, Wx_bf, HIDDEN*INPUT);
    k_cvt<<<(HIDDEN*HIDDEN + 255)/256, 256, 0, stream>>>(Wh, Wh_bf, HIDDEN*HIDDEN);
    k_cvt<<<(HIDDEN*NSYNCH + 255)/256, 256, 0, stream>>>(Wm, Wm_bf, HIDDEN*NSYNCH);
    k_cvt<<<(BATCH*INPUT   + 255)/256, 256, 0, stream>>>(x,  x_bf,  BATCH*INPUT);

    k_init<<<(BH + 255)/256, 256, 0, stream>>>(h_in, a_in, b_in, taup,
                                               out, scal, done, accum);

    // x_drive = x @ Wx^T   (single GEMM, K=512)
    k_gemm<INPUT, false, false><<<gemm_grid, 256, 0, stream>>>(
        x_bf, Wx_bf, nullptr, nullptr, nullptr, nullptr, nullptr,
        x_drive, scal, done, accum);

    for (int i = 0; i < UNFOLDS; ++i) {
        k_sync<<<BH/256, 256, 0, stream>>>(hstate, idxL, idxR, rpar,
                                           alpha, beta, sync_bf, h_bf, done);
        // fused: h@Wh^T + sync@Wm^T + x_drive + b_h -> tanh -> h_update + sum|hu|
        k_gemm<HIDDEN, true, true><<<gemm_grid, 256, 0, stream>>>(
            h_bf, Wh_bf, sync_bf, Wm_bf, x_drive, bh, hstate,
            h_upd, scal, done, accum);
        k_decide<<<1, 32, 0, stream>>>(accum, done, scal, out + 3*(size_t)BH, i);
        k_apply<<<BH/256, 256, 0, stream>>>(hstate, h_upd, scal);
    }
}
